// MultiViewFullBatchGAE_40149354283429
// MI455X (gfx1250) — matrix operand order repaired and checked by reading, not compiled, not run on hardware
//
#include <hip/hip_runtime.h>
#include <hip/hip_bf16.h>
#include <math.h>

typedef __attribute__((ext_vector_type(16))) _Float16 v16h;
typedef __attribute__((ext_vector_type(8)))  float    v8f;

#define WAVE 32

// generic -> LDS(addrspace 3) byte offset, for async-to-LDS VDST operand
__device__ inline unsigned lds_byte_offset(void* p) {
  return (unsigned)(unsigned long long)(__attribute__((address_space(3))) void*)p;
}

// ------------------------------------------------------------------
// convert + zero-pad fp32 [rows,K] -> f16 [rows,Kp]
// ------------------------------------------------------------------
__global__ void k_convert_pad_f16(const float* __restrict__ src, _Float16* __restrict__ dst,
                                  int rows, int K, int Kp) {
  long long idx = (long long)blockIdx.x * blockDim.x + threadIdx.x;
  long long tot = (long long)rows * Kp;
  if (idx >= tot) return;
  int r = (int)(idx / Kp);
  int j = (int)(idx % Kp);
  float v = (j < K) ? src[(long long)r * K + j] : 0.0f;
  dst[idx] = (_Float16)v;
}

// ------------------------------------------------------------------
// acc = (acc + sum_r br[r]) * scale  (+ optional relu)
// ------------------------------------------------------------------
__global__ void k_finalize(float* __restrict__ acc, const float* __restrict__ br,
                           int Nn, int D, int R, float scale, int do_relu) {
  long long idx = (long long)blockIdx.x * blockDim.x + threadIdx.x;
  long long tot = (long long)Nn * D;
  if (idx >= tot) return;
  int j = (int)(idx % D);
  float b = 0.f;
  for (int r = 0; r < R; ++r) b += br[(long long)r * D + j];
  float v = (acc[idx] + b) * scale;
  if (do_relu) v = v > 0.f ? v : 0.f;
  acc[idx] = v;
}

// ------------------------------------------------------------------
// WMMA GEMM with async-LDS A staging:
//   C[M,Nout] = A[M,Kp] (f16) * W[Nout,Kp]^T (f16)  (+bias)
// 256 threads = 8 waves per block. Block owns (8/tilesN) row tiles;
// the A strip (<=32 rows x Kp f16 <= 8KB) is staged into LDS once via
// global_load_async_to_lds_b128 (ASYNCcnt), then every wave computes one
// 16x16 tile reading A fragments from LDS and B fragments from global.
// Fragment layouts used here (wave32; contiguous k, same for A and B):
//   A: lane m=lane&15, khi=lane>>4; half i -> K = 16*khi + i
//   B: lane n=lane&15, khi=lane>>4; half i -> K = 16*khi + i
//   C/D: lane n=lane&15; vgpr j -> row m = j + 8*khi
// ------------------------------------------------------------------
__global__ void k_wmma_gemm(const _Float16* __restrict__ A, const _Float16* __restrict__ W,
                            const float* __restrict__ bias, float* __restrict__ C,
                            int M, int Kp, int Nout) {
  __shared__ _Float16 tileA[32 * 128];       // max strip: 32 rows x Kp=128 (8KB)

  int tilesN = Nout >> 4;                    // 4 or 8
  int rowsPerBlk = (8 / tilesN) * 16;        // 32 or 16
  int row0blk = blockIdx.x * rowsPerBlk;

  // ---- stage A strip into LDS (async, 16B per lane per shot) ----
  int tileBytes = rowsPerBlk * Kp * 2;       // 2KB .. 8KB
  {
    const char* gsrc = (const char*)(A + (long long)row0blk * Kp);
    unsigned lbase = lds_byte_offset(tileA);
    for (int off = (int)threadIdx.x * 16; off < tileBytes; off += (int)blockDim.x * 16) {
      unsigned l = lbase + off;
      unsigned long long g = (unsigned long long)(gsrc + off);
      asm volatile("global_load_async_to_lds_b128 %0, %1, off"
                   :: "v"(l), "v"(g) : "memory");
    }
    asm volatile("s_wait_asynccnt 0" ::: "memory");
  }
  __syncthreads();

  int wv = (int)threadIdx.x >> 5;
  int lane = threadIdx.x & (WAVE - 1);
  int rt = wv / tilesN;                      // row tile within block
  int cn = wv % tilesN;                      // col tile
  int rowL = rt << 4;                        // row offset inside LDS strip
  int col0 = cn << 4;
  int ml = lane & 15;
  int khi = lane >> 4;

  v8f c = {};
  const _Float16* ap = tileA + (rowL + ml) * Kp;              // LDS
  const _Float16* bp = W + (long long)(col0 + ml) * Kp;       // global
  for (int kb = 0; kb < Kp; kb += 32) {
    v16h a, b;
#pragma unroll
    for (int i = 0; i < 16; ++i)
      a[i] = ap[kb + 16 * khi + i];
#pragma unroll
    for (int i = 0; i < 16; ++i)
      b[i] = bp[kb + 16 * khi + i];
    c = __builtin_amdgcn_wmma_f32_16x16x32_f16(false, a, false, b, (short)0, c, false, false);
  }

  float bn = bias ? bias[col0 + ml] : 0.0f;
#pragma unroll
  for (int j = 0; j < 8; ++j) {
    int row = row0blk + rowL + j + 8 * khi;
    if (row < M)
      C[(long long)row * Nout + col0 + ml] = c[j] + bn;
  }
}

// ------------------------------------------------------------------
// COO scatter:  acc[rows[e], :] += vals[e] * t[cols[e], :]
// (dense transform done first via WMMA, so this is the only sparse pass)
// ------------------------------------------------------------------
__global__ void k_spmm_scatter(const int* __restrict__ rows, const int* __restrict__ cols,
                               const float* __restrict__ vals, const float* __restrict__ t,
                               float* __restrict__ acc, int E, int D) {
  long long idx = (long long)blockIdx.x * blockDim.x + threadIdx.x;
  long long tot = (long long)E * D;
  if (idx >= tot) return;
  int e = (int)(idx / D);
  int d = (int)(idx % D);
  float v = vals[e] * t[(long long)cols[e] * D + d];
  atomicAdd(acc + (long long)rows[e] * D + d, v);
}

// ------------------------------------------------------------------
// gated 3-view fusion (softmax over views), L = 64
// ------------------------------------------------------------------
__global__ void k_fuse(const float* __restrict__ zp, const float* __restrict__ zf,
                       const float* __restrict__ zn, const float* __restrict__ gW,
                       const float* __restrict__ gb, float* __restrict__ z, int Nn) {
  int i = blockIdx.x * blockDim.x + threadIdx.x;
  if (i >= Nn) return;
  const float* p = zp + (long long)i * 64;
  const float* f = zf + (long long)i * 64;
  const float* n = zn + (long long)i * 64;
  float s0 = gb[0], s1 = gb[1], s2 = gb[2];
  for (int c = 0; c < 64; ++c) {
    s0 += p[c] * gW[c];
    s1 += f[c] * gW[64 + c];
    s2 += n[c] * gW[128 + c];
  }
  float m = fmaxf(s0, fmaxf(s1, s2));
  float e0 = expf(s0 - m), e1 = expf(s1 - m), e2 = expf(s2 - m);
  float inv = 1.0f / (e0 + e1 + e2);
  float a0 = e0 * inv, a1 = e1 * inv, a2 = e2 * inv;
  float* zo = z + (long long)i * 64;
  for (int c = 0; c < 64; ++c)
    zo[c] = a0 * p[c] + a1 * f[c] + a2 * n[c];
}

// ------------------------------------------------------------------
// edge decoder: 16 edges per block, feats [16 x 256] f16 staged in LDS,
// 4 waves x (16x16 WMMA tile, K=256) cover DH=64 hidden units,
// relu + dot with dec_W2 reduced via ds_add_f32.
// ------------------------------------------------------------------
__global__ void k_decoder(const float* __restrict__ z, const int* __restrict__ esrc,
                          const int* __restrict__ edst, const _Float16* __restrict__ W1h,
                          const float* __restrict__ b1, const float* __restrict__ W2,
                          const float* __restrict__ b2, float* __restrict__ out, int ED) {
  __shared__ _Float16 feats[16][256];
  __shared__ float partial[16];
  int tid = threadIdx.x;            // 128 threads
  int e0  = blockIdx.x * 16;
  {
    int el   = tid >> 3;            // edge in tile 0..15
    int cseg = (tid & 7) * 8;       // 8 latent channels per thread
    int e = e0 + el;
    if (e < ED) {
      const float* zs = z + (long long)esrc[e] * 64;
      const float* zd = z + (long long)edst[e] * 64;
      for (int c = cseg; c < cseg + 8; ++c) {
        float a = zs[c], b = zd[c];
        feats[el][c]       = (_Float16)a;
        feats[el][64 + c]  = (_Float16)b;
        feats[el][128 + c] = (_Float16)(a * b);
        feats[el][192 + c] = (_Float16)fabsf(a - b);
      }
    } else {
      for (int c = cseg; c < cseg + 8; ++c) {
        feats[el][c] = (_Float16)0.f; feats[el][64 + c] = (_Float16)0.f;
        feats[el][128 + c] = (_Float16)0.f; feats[el][192 + c] = (_Float16)0.f;
      }
    }
    if (tid < 16) partial[tid] = 0.f;
  }
  __syncthreads();

  int wave = tid >> 5;              // 4 waves -> hidden cols [16*wave, 16*wave+16)
  int lane = tid & 31;
  int col0 = wave * 16;
  int ml = lane & 15, khi = lane >> 4;
  v8f c = {};
  const _Float16* bp = W1h + (long long)(col0 + ml) * 256;
  for (int kb = 0; kb < 256; kb += 32) {
    v16h a, b;
#pragma unroll
    for (int i = 0; i < 16; ++i)
      a[i] = feats[ml][kb + 16 * khi + i];
#pragma unroll
    for (int i = 0; i < 16; ++i)
      b[i] = bp[kb + 16 * khi + i];
    c = __builtin_amdgcn_wmma_f32_16x16x32_f16(false, a, false, b, (short)0, c, false, false);
  }
  int n = col0 + ml;
  float bn = b1[n], wn = W2[n];
#pragma unroll
  for (int j = 0; j < 8; ++j) {
    int row = j + 8 * khi;          // edge index within tile
    float h = c[j] + bn;
    h = h > 0.f ? h : 0.f;
    atomicAdd(&partial[row], h * wn);
  }
  __syncthreads();
  if (tid < 16 && (e0 + tid) < ED)
    out[e0 + tid] = partial[tid] + b2[0];
}

// ==================================================================
// host orchestration
// ==================================================================
static inline int cdiv_i(long long a, long long b) { return (int)((a + b - 1) / b); }

static void launch_convert(const float* src, _Float16* dst, int rows, int K, int Kp,
                           hipStream_t s) {
  long long tot = (long long)rows * Kp;
  k_convert_pad_f16<<<cdiv_i(tot, 256), 256, 0, s>>>(src, dst, rows, K, Kp);
}

static void launch_gemm(const _Float16* A, const _Float16* W, const float* bias, float* C,
                        int M, int Kp, int Nout, hipStream_t s) {
  int tilesN = Nout >> 4;
  int rowsPerBlk = (8 / tilesN) * 16;
  k_wmma_gemm<<<cdiv_i(M, rowsPerBlk), 256, 0, s>>>(A, W, bias, C, M, Kp, Nout);
}

// one RGCN layer: acc = x@Ws.T + bs + sum_r spmm_r(x@Wr.T) (+br), /5, relu?
static void rgcn_layer(const _Float16* xf16, int Kp, int din, int dout, int Nn,
                       const float* Ws, const float* bs, const float* Wr, const float* br,
                       const int* arows, const int* acols, const float* avals, int E,
                       _Float16* wbuf, float* acc, float* tmp, int do_relu, hipStream_t s) {
  const int SLOT = 16384; // _Float16 elems per weight slot (32KB)
  // convert weights (zero-padded K)
  launch_convert(Ws, wbuf, dout, din, Kp, s);
  for (int r = 0; r < 4; ++r)
    launch_convert(Wr + (long long)r * dout * din, wbuf + (long long)(r + 1) * SLOT,
                   dout, din, Kp, s);
  // self transform (overwrites acc, adds bs)
  launch_gemm(xf16, wbuf, bs, acc, Nn, Kp, dout, s);
  // relation transforms + scatter
  for (int r = 0; r < 4; ++r) {
    launch_gemm(xf16, wbuf + (long long)(r + 1) * SLOT, nullptr, tmp, Nn, Kp, dout, s);
    long long tot = (long long)E * dout;
    k_spmm_scatter<<<cdiv_i(tot, 256), 256, 0, s>>>(arows + (long long)r * E,
                                                    acols + (long long)r * E,
                                                    avals + (long long)r * E,
                                                    tmp, acc, E, dout);
  }
  long long tot = (long long)Nn * dout;
  k_finalize<<<cdiv_i(tot, 256), 256, 0, s>>>(acc, br, Nn, dout, 4, 0.2f, do_relu);
}

extern "C" void kernel_launch(void* const* d_in, const int* in_sizes, int n_in,
                              void* d_out, int out_size, void* d_ws, size_t ws_size,
                              hipStream_t stream) {
  (void)n_in; (void)ws_size;
  const float* x_p = (const float*)d_in[0];
  const float* x_f = (const float*)d_in[1];
  const float* x_n = (const float*)d_in[2];
  // per-view weights: [Ws, bs, Wr, br] x {layer1, layer2}
  const float* W[3][8];
  for (int v = 0; v < 3; ++v)
    for (int k = 0; k < 8; ++k) W[v][k] = (const float*)d_in[3 + v * 8 + k];
  const float* gate_W = (const float*)d_in[27];
  const float* gate_b = (const float*)d_in[28];
  const float* dec_W1 = (const float*)d_in[29];
  const float* dec_b1 = (const float*)d_in[30];
  const float* dec_W2 = (const float*)d_in[31];
  const float* dec_b2 = (const float*)d_in[32];
  const int*   adj_rows = (const int*)d_in[33];
  const int*   adj_cols = (const int*)d_in[34];
  const float* adj_vals = (const float*)d_in[35];
  const int*   edge_src = (const int*)d_in[36];
  const int*   edge_dst = (const int*)d_in[37];

  const int Nn = in_sizes[0] / 62;       // 50000
  const int E  = in_sizes[33] / 4;       // 800000
  const int ED = in_sizes[36];           // 200000
  const int H = 128, L = 64;
  const int dins[3] = {62, 37, 37};
  const float* xs[3] = {x_p, x_f, x_n};

  // workspace layout (256B-aligned by construction)
  char* ws = (char*)d_ws;
  _Float16* xf16  = (_Float16*)(ws + 0);                       // [N,64]  f16
  _Float16* hf16  = (_Float16*)(ws + 6400000LL);               // [N,128] f16
  float*    acc   = (float*)(ws + 19200000LL);                 // [N,128] f32
  float*    tmp   = (float*)(ws + 44800000LL);                 // [N,128] f32
  float*    zv[3] = {(float*)(ws + 70400000LL),                // z_p [N,64]
                     (float*)(ws + 83200000LL),                // z_f
                     (float*)(ws + 96000000LL)};               // z_n
  float*    zfu   = (float*)(ws + 108800000LL);                // fused z [N,64]
  _Float16* wbuf  = (_Float16*)(ws + 121600000LL);             // 5 x 32KB weight slots

  for (int v = 0; v < 3; ++v) {
    // layer 1: din -> H=128, relu, pad K to 64
    launch_convert(xs[v], xf16, Nn, dins[v], 64, stream);
    rgcn_layer(xf16, 64, dins[v], H, Nn, W[v][0], W[v][1], W[v][2], W[v][3],
               adj_rows, adj_cols, adj_vals, E, wbuf, acc, tmp, /*relu=*/1, stream);
    // layer 2: H=128 -> L=64, no relu
    launch_convert(acc, hf16, Nn, H, H, stream);
    rgcn_layer(hf16, H, H, L, Nn, W[v][4], W[v][5], W[v][6], W[v][7],
               adj_rows, adj_cols, adj_vals, E, wbuf, zv[v], tmp, /*relu=*/0, stream);
  }

  // gated fusion
  k_fuse<<<cdiv_i(Nn, 256), 256, 0, stream>>>(zv[0], zv[1], zv[2], gate_W, gate_b, zfu, Nn);

  // decoder: convert W1 [64,256] to f16, then WMMA decode 16 edges/block
  launch_convert(dec_W1, wbuf, 64, 256, 256, stream);
  k_decoder<<<cdiv_i(ED, 16), 128, 0, stream>>>(zfu, edge_src, edge_dst, wbuf,
                                                dec_b1, dec_W2, dec_b2,
                                                (float*)d_out, ED);
  (void)out_size;
}
